// QConv_75445395522273
// MI455X (gfx1250) — compile-verified, matrix operand order checked by reading, not run on hardware
//
#include <hip/hip_runtime.h>

// QConv (graph conv) for MI455X / gfx1250, wave32 + WMMA f16->f32.
//
// Pipeline (all on `stream`, scratch in d_ws):
//   zero_kernel : clear sums[N,64] and deg[N]
//   cvt_kernel  : W1 (64x67 f32) -> W1h (64x96 f16, zero-padded),
//                 W2 (64x128 f32) -> W2h (64x128 f16)
//   deg_kernel  : deg[dst[e]] += 1 (global_atomic_add_f32)
//   edge_kernel : per 16-edge tile per wave: K padded 67->96,
//                 A = f16([h[src] | w | 0]) 16x96, 3 WMMA per N-tile (12 total),
//                 leaky_relu, branch-free scatter atomicAdd into sums[dst]
//                 (sums is 12.8 MB -> L2-resident on the 192 MB L2).
//   node_kernel : per 16-node tile per wave:
//                 A = f16([h | sums/max(deg,1)]) 16x128, 4 WMMA per N-tile (16),
//                 + b2, branch-free store.

typedef __attribute__((ext_vector_type(16))) _Float16 v16h;
typedef __attribute__((ext_vector_type(8)))  float    v8f;

#define N_NODES_C   50000
#define N_EDGES_C   1600000
#define IN_FEAT_C   64
#define INTER_C     64
#define OUT_C       64
#define EWD_C       3
#define W1_LDK      96      // f16 row stride for padded W1 (67 -> 96: full K chunk 2)
#define W2_LDK      128
#define NEG_SLOPE_C 0.01f

// ---- fragment helpers ------------------------------------------------------
// A-matrix 16x32 f16 layout (ISA 7.12.2): lane L (row M = L&15), hsel = L>>4:
//   half i in [0,8)  -> K = 32*kc + 8*hsel + i
//   half i in [8,16) -> K = 32*kc + 16 + 8*hsel + (i-8)
// i.e. two runs of 8 contiguous K values -> two float4-pairs per fragment.

__device__ __forceinline__ void pack8(v16h& a, int off, float4 x, float4 y, float s) {
  a[off + 0] = (_Float16)(x.x * s); a[off + 1] = (_Float16)(x.y * s);
  a[off + 2] = (_Float16)(x.z * s); a[off + 3] = (_Float16)(x.w * s);
  a[off + 4] = (_Float16)(y.x * s); a[off + 5] = (_Float16)(y.y * s);
  a[off + 6] = (_Float16)(y.z * s); a[off + 7] = (_Float16)(y.w * s);
}

__device__ __forceinline__ v16h make_afrag(const float4* __restrict__ v, int kc,
                                           int hsel, float s) {
  const int b = 8 * kc + 2 * hsel;       // float4 index of first 8-float run
  v16h a;
  pack8(a, 0, v[b],     v[b + 1], s);    // K = 32kc + 8h .. +7
  pack8(a, 8, v[b + 4], v[b + 5], s);    // K = 32kc + 16 + 8h .. +7
  return a;
}

// B-matrix 32x16 f16 fragment for t = A @ W^T: B[k][n] = W[n][k] (row-major W).
// Mirrors the A layout with N across lanes; two 8-half contiguous runs in k.
__device__ __forceinline__ v16h load_bfrag(const _Float16* __restrict__ Wh, int ldk,
                                           int ntile, int kc, int lane) {
  const int n     = (lane & 15) + (ntile << 4);
  const int kbase = (kc << 5) + ((lane >> 4) << 3);
  const _Float16* p = Wh + (long)n * ldk + kbase;
  v16h b;
#pragma unroll
  for (int i = 0; i < 8; ++i) { b[i] = p[i]; b[8 + i] = p[16 + i]; }
  return b;
}

__device__ __forceinline__ float leaky(float x) {
  return fmaxf(x, 0.0f) + NEG_SLOPE_C * fminf(x, 0.0f);
}

// ---- setup kernels ---------------------------------------------------------

__global__ void zero_kernel(float* __restrict__ p, int n) {
  int i = blockIdx.x * blockDim.x + threadIdx.x;
  if (i < n) p[i] = 0.0f;
}

__global__ void cvt_kernel(const float* __restrict__ W1, const float* __restrict__ W2,
                           _Float16* __restrict__ W1h, _Float16* __restrict__ W2h) {
  int i = blockIdx.x * blockDim.x + threadIdx.x;
  if (i < 64 * W1_LDK) {
    int j = i / W1_LDK, k = i % W1_LDK;
    W1h[i] = (k < (IN_FEAT_C + EWD_C)) ? (_Float16)W1[j * (IN_FEAT_C + EWD_C) + k]
                                       : (_Float16)0.0f;
  }
  if (i < 64 * W2_LDK) W2h[i] = (_Float16)W2[i];
}

__global__ void deg_kernel(const int* __restrict__ dst, float* __restrict__ deg) {
  int i = blockIdx.x * blockDim.x + threadIdx.x;
  if (i < N_EDGES_C) atomicAdd(&deg[dst[i]], 1.0f);
}

// ---- edge kernel: fused gather + linear1 + leaky_relu + segment-sum --------
// Grid sized exactly: N_EDGES is a multiple of 16*8, so no row guards needed.

__global__ __launch_bounds__(256) void edge_kernel(
    const float* __restrict__ h, const float* __restrict__ w,
    const int* __restrict__ src, const int* __restrict__ dst,
    const _Float16* __restrict__ W1h, float* __restrict__ sums) {
  const int lane = threadIdx.x & 31;
  const int wave = threadIdx.x >> 5;
  const int tile = blockIdx.x * 8 + wave;     // one 16-edge tile per wave
  const int e0   = tile << 4;
  if (e0 >= N_EDGES_C) return;                // wave-uniform (never taken)

  const int mrow = lane & 15;
  const int hsel = lane >> 4;
  const int erow = e0 + mrow;
  const int s    = src[erow];

  // A fragments: 16 edges x K96 = [h[src] (0..63) | w (64..66) | zeros].
  const float4* hv = (const float4*)(h + (long)s * IN_FEAT_C);
  const v16h a0 = make_afrag(hv, 0, hsel, 1.0f);   // K  0..31
  const v16h a1 = make_afrag(hv, 1, hsel, 1.0f);   // K 32..63
  v16h a2 = {};                                    // K 64..95 (edge-w dims)
  {
    // K=64,65,66 live in halfs 0..2 of the hsel==0 lanes; everything else 0.
    const float m = (hsel == 0) ? 1.0f : 0.0f;
    a2[0] = (_Float16)(m * w[(long)erow * EWD_C + 0]);
    a2[1] = (_Float16)(m * w[(long)erow * EWD_C + 1]);
    a2[2] = (_Float16)(m * w[(long)erow * EWD_C + 2]);
  }

  // dst node for each accumulator row r: address is uniform across each
  // half-wave -> 8 broadcast L0 hits, reused by all 4 N-tiles.
  int dr[8];
#pragma unroll
  for (int r = 0; r < 8; ++r) dr[r] = dst[e0 + r + (hsel << 3)];

#pragma unroll
  for (int nt = 0; nt < 4; ++nt) {
    const int n = mrow + (nt << 4);            // output column (inter dim)
    v8f c = {};
    c = __builtin_amdgcn_wmma_f32_16x16x32_f16(
        false, a0, false, load_bfrag(W1h, W1_LDK, nt, 0, lane), (short)0, c, false, false);
    c = __builtin_amdgcn_wmma_f32_16x16x32_f16(
        false, a1, false, load_bfrag(W1h, W1_LDK, nt, 1, lane), (short)0, c, false, false);
    c = __builtin_amdgcn_wmma_f32_16x16x32_f16(
        false, a2, false, load_bfrag(W1h, W1_LDK, nt, 2, lane), (short)0, c, false, false);
#pragma unroll
    for (int r = 0; r < 8; ++r) {
      const float v = leaky(c[r]);
      atomicAdd(&sums[(long)dr[r] * INTER_C + n], v);   // branch-free scatter
    }
  }
}

// ---- node kernel: [h | sums/deg] @ W2^T + b2 -------------------------------
// N_NODES = 3125 * 16 exactly -> no row guards; tile guard is wave-uniform.

__global__ __launch_bounds__(256) void node_kernel(
    const float* __restrict__ h, const float* __restrict__ sums,
    const float* __restrict__ deg, const _Float16* __restrict__ W2h,
    const float* __restrict__ b2, float* __restrict__ out) {
  const int lane = threadIdx.x & 31;
  const int wave = threadIdx.x >> 5;
  const int tile = blockIdx.x * 8 + wave;     // one 16-node tile per wave
  const int n0   = tile << 4;
  if (n0 >= N_NODES_C) return;                // wave-uniform (last block only)

  const int mrow = lane & 15;
  const int hsel = lane >> 4;
  const int node = n0 + mrow;
  const float rden = 1.0f / fmaxf(deg[node], 1.0f);   // segment mean

  const float4* hv = (const float4*)(h    + (long)node * IN_FEAT_C);
  const float4* sv = (const float4*)(sums + (long)node * INTER_C);
  v16h a[4];
  a[0] = make_afrag(hv, 0, hsel, 1.0f);   // K   0.. 63 : h
  a[1] = make_afrag(hv, 1, hsel, 1.0f);
  a[2] = make_afrag(sv, 0, hsel, rden);   // K  64..127 : h_N = sums/deg
  a[3] = make_afrag(sv, 1, hsel, rden);

#pragma unroll
  for (int nt = 0; nt < 4; ++nt) {
    const int n = mrow + (nt << 4);
    const float bias = b2[n];
    v8f c;
#pragma unroll
    for (int r = 0; r < 8; ++r) c[r] = bias;
#pragma unroll
    for (int kc = 0; kc < 4; ++kc)
      c = __builtin_amdgcn_wmma_f32_16x16x32_f16(
          false, a[kc], false, load_bfrag(W2h, W2_LDK, nt, kc, lane),
          (short)0, c, false, false);
#pragma unroll
    for (int r = 0; r < 8; ++r) {
      const int row = n0 + r + (hsel << 3);
      out[(long)row * OUT_C + n] = c[r];      // branch-free store
    }
  }
}

// ---- launcher --------------------------------------------------------------

extern "C" void kernel_launch(void* const* d_in, const int* in_sizes, int n_in,
                              void* d_out, int out_size, void* d_ws, size_t ws_size,
                              hipStream_t stream) {
  const float* h   = (const float*)d_in[0];
  const float* w   = (const float*)d_in[1];
  const int*   src = (const int*)d_in[2];
  const int*   dst = (const int*)d_in[3];
  const float* W1  = (const float*)d_in[4];
  const float* W2  = (const float*)d_in[5];
  const float* b2  = (const float*)d_in[6];
  float* out = (float*)d_out;

  // Workspace layout (~13.04 MB): sums | deg | W1h | W2h  (all 16B aligned)
  float*    sums = (float*)d_ws;
  float*    deg  = sums + (size_t)N_NODES_C * INTER_C;
  _Float16* W1h  = (_Float16*)(deg + N_NODES_C);
  _Float16* W2h  = W1h + 64 * W1_LDK;

  const int nz = N_NODES_C * (INTER_C + 1);
  zero_kernel<<<(nz + 255) / 256, 256, 0, stream>>>(sums, nz);
  cvt_kernel<<<(64 * W2_LDK + 255) / 256, 256, 0, stream>>>(W1, W2, W1h, W2h);
  deg_kernel<<<(N_EDGES_C + 255) / 256, 256, 0, stream>>>(dst, deg);

  const int etiles = (N_EDGES_C + 15) / 16;   // 100000, 8 waves/block -> 12500
  edge_kernel<<<(etiles + 7) / 8, 256, 0, stream>>>(h, w, src, dst, W1h, sums);

  const int ntiles = (N_NODES_C + 15) / 16;   // 3125 -> 391 blocks
  node_kernel<<<(ntiles + 7) / 8, 256, 0, stream>>>(h, sums, deg, W2h, b2, out);
}
